// PEBlock_27771258536250
// MI455X (gfx1250) — matrix operand order repaired and checked by reading, not compiled, not run on hardware
//
#include <hip/hip_runtime.h>

typedef __attribute__((ext_vector_type(16))) _Float16 v16h;
typedef __attribute__((ext_vector_type(8)))  _Float16 v8h;
typedef __attribute__((ext_vector_type(8)))  float    v8f;

#define LRELU(x) ((x) >= 0.f ? (x) : 0.1f * (x))

// Build A fragment (16x32 f16): lane L (L<16): row M=L, K=0..15
// lane L+16: row M=L, K=16..31.  rowp points at (row, kBase) in LDS.
__device__ __forceinline__ v16h mk_a_frag(const _Float16* rowp, int grp) {
    v8h lo = *(const v8h*)(rowp + 16 * grp);
    v8h hi = *(const v8h*)(rowp + 16 * grp + 8);
    v16h a;
#pragma unroll
    for (int i = 0; i < 8; ++i) { a[i] = lo[i]; a[8 + i] = hi[i]; }
    return a;
}

// ---------------------------------------------------------------------------
// Fused encoder block: y = lrelu( bn(lrelu(x@W1+b1)) @ W2 + b2 )
// K1 = padded input width (32 for 9-ch level 0, 64 for the rest).
// FIXUP converts -inf sentinel (empty seg-max segments) to 0 on load.
// ---------------------------------------------------------------------------
template <int K1, bool FIXUP>
__launch_bounds__(256)
__global__ void enc_kernel(const float* __restrict__ x, int n, int inK,
                           const float* __restrict__ W1, const float* __restrict__ b1,
                           const float* __restrict__ bn_g, const float* __restrict__ bn_b,
                           const float* __restrict__ bn_m, const float* __restrict__ bn_v,
                           const float* __restrict__ W2, const float* __restrict__ b2,
                           float* __restrict__ out) {
    constexpr int H = 32, C = 64, ROWS = 128;
    __shared__ __align__(32) _Float16 sX[ROWS * K1];   // activations, f16, zero padded
    __shared__ __align__(32) _Float16 sW1t[H * K1];    // W1 transposed [H][K1]
    __shared__ __align__(32) _Float16 sW2t[C * H];     // W2 transposed [C][H]
    __shared__ __align__(32) _Float16 sH[ROWS * H];    // hidden activations

    const int tid = threadIdx.x;
    const int rowBase = blockIdx.x * ROWS;

    for (int i = tid; i < ROWS * K1; i += 256) {
        int r = i / K1, k = i - r * K1;
        int row = rowBase + r;
        float val = 0.f;
        if (row < n && k < inK) {
            val = x[(long)row * inK + k];
            if (FIXUP && __float_as_uint(val) == 0xFF800000u) val = 0.f;
        }
        sX[i] = (_Float16)val;
    }
    for (int i = tid; i < H * K1; i += 256) {
        int nh = i / K1, k = i - nh * K1;
        sW1t[i] = (_Float16)((k < inK) ? W1[k * H + nh] : 0.f);
    }
    for (int i = tid; i < C * H; i += 256) {
        int nc = i / H, k = i - nc * H;
        sW2t[i] = (_Float16)W2[k * C + nc];
    }
    __syncthreads();

    const int lane = tid & 31;
    const int wave = tid >> 5;
    const int ln15 = lane & 15;
    const int grp  = lane >> 4;

    // ---- layer 1: [16 x K1] @ [K1 x 32] ----
    v16h aF[K1 / 32];
#pragma unroll
    for (int c = 0; c < K1 / 32; ++c)
        aF[c] = mk_a_frag(&sX[(wave * 16 + ln15) * K1 + c * 32], grp);

#pragma unroll
    for (int t = 0; t < H / 16; ++t) {
        v8f acc = {};
#pragma unroll
        for (int c = 0; c < K1 / 32; ++c) {
            v16h bF = *(const v16h*)&sW1t[(t * 16 + ln15) * K1 + c * 32 + 16 * grp];
            acc = __builtin_amdgcn_wmma_f32_16x16x32_f16(false, aF[c], false, bF,
                                                         (short)0, acc, false, false);
        }
        int col = t * 16 + ln15;
        float bb = b1[col];
        float sc = bn_g[col] * rsqrtf(bn_v[col] + 1e-5f);
        float sh = bn_b[col] - bn_m[col] * sc;
#pragma unroll
        for (int r = 0; r < 8; ++r) {
            float h = acc[r] + bb;        // C/D layout: lane=col, VGPR r -> row grp*8+r
            h = LRELU(h);
            h = h * sc + sh;
            sH[(wave * 16 + grp * 8 + r) * H + col] = (_Float16)h;
        }
    }
    __syncthreads();

    // ---- layer 2: [16 x 32] @ [32 x 64] ----
    v16h aH = mk_a_frag(&sH[(wave * 16 + ln15) * H], grp);
#pragma unroll
    for (int t = 0; t < C / 16; ++t) {
        v8f acc = {};
        v16h bF = *(const v16h*)&sW2t[(t * 16 + ln15) * H + 16 * grp];
        acc = __builtin_amdgcn_wmma_f32_16x16x32_f16(false, aH, false, bF,
                                                     (short)0, acc, false, false);
        int col = t * 16 + ln15;
        float bb = b2[col];
#pragma unroll
        for (int r = 0; r < 8; ++r) {
            int row = rowBase + wave * 16 + grp * 8 + r;
            if (row < n) {
                float y = acc[r] + bb;
                out[(long)row * C + col] = LRELU(y);
            }
        }
    }
}

// ---------------------------------------------------------------------------
// Fused decoder block:
//   skip = lrelu(cur @ mlpW + mlpb)
//   h    = concat(skip, last) @ decW + decb   (= skip@Wa + last@Wb + decb)
//   y    = lrelu(bn(h))
// last = prev[inv[row]] (inv==nullptr -> identity). In-place safe (out may
// alias cur): each block reads only its own rows into LDS before writing.
// ---------------------------------------------------------------------------
__launch_bounds__(256)
__global__ void dec_kernel(const float* __restrict__ cur, int n,
                           const float* __restrict__ prev, const int* __restrict__ inv,
                           const float* __restrict__ mlpW, const float* __restrict__ mlpb,
                           const float* __restrict__ decW, const float* __restrict__ decb,
                           const float* __restrict__ bn_g, const float* __restrict__ bn_b,
                           const float* __restrict__ bn_m, const float* __restrict__ bn_v,
                           float* __restrict__ out) {
    constexpr int C = 64, ROWS = 128;
    __shared__ __align__(32) _Float16 sCur[ROWS * C];   // reused for skip
    __shared__ __align__(32) _Float16 sLast[ROWS * C];
    __shared__ __align__(32) _Float16 sMt[C * C];       // mlpW transposed
    __shared__ __align__(32) _Float16 sWaT[C * C];      // decW[0:64]  transposed
    __shared__ __align__(32) _Float16 sWbT[C * C];      // decW[64:128] transposed

    const int tid = threadIdx.x;
    const int rowBase = blockIdx.x * ROWS;

    for (int i = tid; i < ROWS * C; i += 256) {
        int r = i >> 6, k = i & 63;
        int row = rowBase + r;
        float cv = 0.f, lv = 0.f;
        if (row < n) {
            cv = cur[(long)row * C + k];
            int src = inv ? inv[row] : row;
            lv = prev[(long)src * C + k];
        }
        sCur[i]  = (_Float16)cv;
        sLast[i] = (_Float16)lv;
    }
    for (int i = tid; i < C * C; i += 256) {
        int nc = i >> 6, k = i & 63;
        sMt[i]  = (_Float16)mlpW[k * C + nc];
        sWaT[i] = (_Float16)decW[k * C + nc];
        sWbT[i] = (_Float16)decW[(C + k) * C + nc];
    }
    __syncthreads();

    const int lane = tid & 31, wave = tid >> 5;
    const int ln15 = lane & 15, grp = lane >> 4;

    v16h aC[2], aL[2];
#pragma unroll
    for (int c = 0; c < 2; ++c) {
        aC[c] = mk_a_frag(&sCur[(wave * 16 + ln15) * C + c * 32], grp);
        aL[c] = mk_a_frag(&sLast[(wave * 16 + ln15) * C + c * 32], grp);
    }

    // ---- skip = lrelu(cur @ mlpW + mlpb), overwrite this wave's sCur rows ----
#pragma unroll
    for (int t = 0; t < 4; ++t) {
        v8f acc = {};
#pragma unroll
        for (int c = 0; c < 2; ++c) {
            v16h bF = *(const v16h*)&sMt[(t * 16 + ln15) * C + c * 32 + 16 * grp];
            acc = __builtin_amdgcn_wmma_f32_16x16x32_f16(false, aC[c], false, bF,
                                                         (short)0, acc, false, false);
        }
        int col = t * 16 + ln15;
        float bb = mlpb[col];
#pragma unroll
        for (int r = 0; r < 8; ++r) {
            float s = LRELU(acc[r] + bb);
            sCur[(wave * 16 + grp * 8 + r) * C + col] = (_Float16)s;
        }
    }
    __syncthreads();

    v16h aS[2];
#pragma unroll
    for (int c = 0; c < 2; ++c)
        aS[c] = mk_a_frag(&sCur[(wave * 16 + ln15) * C + c * 32], grp);

    // ---- h = skip@Wa + last@Wb ; y = lrelu(bn(h + decb)) ----
#pragma unroll
    for (int t = 0; t < 4; ++t) {
        v8f acc = {};
#pragma unroll
        for (int c = 0; c < 2; ++c) {
            v16h bA = *(const v16h*)&sWaT[(t * 16 + ln15) * C + c * 32 + 16 * grp];
            acc = __builtin_amdgcn_wmma_f32_16x16x32_f16(false, aS[c], false, bA,
                                                         (short)0, acc, false, false);
            v16h bB = *(const v16h*)&sWbT[(t * 16 + ln15) * C + c * 32 + 16 * grp];
            acc = __builtin_amdgcn_wmma_f32_16x16x32_f16(false, aL[c], false, bB,
                                                         (short)0, acc, false, false);
        }
        int col = t * 16 + ln15;
        float bb = decb[col];
        float sc = bn_g[col] * rsqrtf(bn_v[col] + 1e-5f);
        float sh = bn_b[col] - bn_m[col] * sc;
#pragma unroll
        for (int r = 0; r < 8; ++r) {
            int row = rowBase + wave * 16 + grp * 8 + r;
            if (row < n) {
                float h = (acc[r] + bb) * sc + sh;
                out[(long)row * C + col] = LRELU(h);
            }
        }
    }
}

// ---------------------------------------------------------------------------
// segment-max: init to -inf, bit-trick atomic max, consumer maps -inf -> 0
// ---------------------------------------------------------------------------
__global__ void fill_kernel(float* p, long count, unsigned int bits) {
    long i = (long)blockIdx.x * blockDim.x + threadIdx.x;
    if (i < count) p[i] = __uint_as_float(bits);
}

__device__ __forceinline__ void atomicMaxF(float* addr, float val) {
    if (val >= 0.f) atomicMax((int*)addr, __float_as_int(val));
    else            atomicMin((unsigned int*)addr, __float_as_uint(val));
}

__global__ void segmax_kernel(const float* __restrict__ src, const int* __restrict__ inv,
                              float* __restrict__ dst, int nsrc) {
    long i = (long)blockIdx.x * blockDim.x + threadIdx.x;
    if (i < (long)nsrc * 64) {
        int r = (int)(i >> 6), c = (int)(i & 63);
        atomicMaxF(&dst[(long)inv[r] * 64 + c], src[i]);
    }
}

// ---------------------------------------------------------------------------
extern "C" void kernel_launch(void* const* d_in, const int* in_sizes, int n_in,
                              void* d_out, int out_size, void* d_ws, size_t ws_size,
                              hipStream_t stream) {
    constexpr int C = 64, H = 32;
    const float* pt_fea = (const float*)d_in[0];
    const int* inv2  = (const int*)d_in[1];
    const int* inv4  = (const int*)d_in[2];
    const int* inv8  = (const int*)d_in[3];
    const int* inv16 = (const int*)d_in[4];
    const int N1 = in_sizes[1], N2 = in_sizes[2], N4 = in_sizes[3], N8 = in_sizes[4];
    const int inCh = in_sizes[0] / N1;
    const int N16 = (int)((long)out_size / C - (long)N1 - N2 - N4 - N8);

    const float* e0W1 = (const float*)d_in[5];
    const float* e0b1 = (const float*)d_in[6];
    const float* e0g  = (const float*)d_in[7];
    const float* e0be = (const float*)d_in[8];
    const float* e0m  = (const float*)d_in[9];
    const float* e0v  = (const float*)d_in[10];
    const float* e0W2 = (const float*)d_in[11];
    const float* e0b2 = (const float*)d_in[12];
    const float* rW1  = (const float*)d_in[13];
    const float* rb1  = (const float*)d_in[14];
    const float* rg   = (const float*)d_in[15];
    const float* rbe  = (const float*)d_in[16];
    const float* rm   = (const float*)d_in[17];
    const float* rv   = (const float*)d_in[18];
    const float* rW2  = (const float*)d_in[19];
    const float* rb2  = (const float*)d_in[20];
    const float* mW   = (const float*)d_in[21];
    const float* mb   = (const float*)d_in[22];
    const float* dW   = (const float*)d_in[23];
    const float* db   = (const float*)d_in[24];
    const float* dg   = (const float*)d_in[25];
    const float* dbe  = (const float*)d_in[26];
    const float* dm   = (const float*)d_in[27];
    const float* dv   = (const float*)d_in[28];

    // d_out layout: p1f | p2f | p3f | p4f | p5f  (we write p1..p5 here first,
    // then the decoder overwrites each level in place — race-free, see kernel).
    float* p1 = (float*)d_out;
    float* p2 = p1 + (long)N1 * C;
    float* p3 = p2 + (long)N2 * C;
    float* p4 = p3 + (long)N4 * C;
    float* p5 = p4 + (long)N8 * C;
    float* obuf = (float*)d_ws;   // seg-max scratch, max N2*C floats

    auto grows = [](int n) { return dim3((unsigned)((n + 127) / 128)); };
    auto g256  = [](long c) { return dim3((unsigned)((c + 255) / 256)); };

    // ---- encoder pyramid ----
    enc_kernel<32, false><<<grows(N1), 256, 0, stream>>>(
        pt_fea, N1, inCh, e0W1, e0b1, e0g, e0be, e0m, e0v, e0W2, e0b2, p1);

    const int  Ns[4]   = {N2, N4, N8, N16};
    const int  Np[4]   = {N1, N2, N4, N8};
    const int* invs[4] = {inv2, inv4, inv8, inv16};
    const float* srcs[4] = {p1, p2, p3, p4};
    float* dsts[4]       = {p2, p3, p4, p5};
    for (int i = 0; i < 4; ++i) {
        long oc = (long)Ns[i] * C;
        fill_kernel<<<g256(oc), 256, 0, stream>>>(obuf, oc, 0xFF800000u);
        segmax_kernel<<<g256((long)Np[i] * C), 256, 0, stream>>>(srcs[i], invs[i], obuf, Np[i]);
        enc_kernel<64, true><<<grows(Ns[i]), 256, 0, stream>>>(
            obuf, Ns[i], C,
            rW1 + (long)i * C * H, rb1 + i * H, rg + i * H, rbe + i * H,
            rm + i * H, rv + i * H, rW2 + (long)i * H * C, rb2 + i * C, dsts[i]);
    }

    // ---- decoder (in place: p5->p5f, p4->p4f, ...) ----
    dec_kernel<<<grows(N16), 256, 0, stream>>>(
        p5, N16, p5, nullptr,
        mW, mb, dW, db, dg, dbe, dm, dv, p5);
    dec_kernel<<<grows(N8), 256, 0, stream>>>(
        p4, N8, p5, inv16,
        mW + 1L * C * C, mb + 1 * C, dW + 1L * 2 * C * C, db + 1 * C,
        dg + 1 * C, dbe + 1 * C, dm + 1 * C, dv + 1 * C, p4);
    dec_kernel<<<grows(N4), 256, 0, stream>>>(
        p3, N4, p4, inv8,
        mW + 2L * C * C, mb + 2 * C, dW + 2L * 2 * C * C, db + 2 * C,
        dg + 2 * C, dbe + 2 * C, dm + 2 * C, dv + 2 * C, p3);
    dec_kernel<<<grows(N2), 256, 0, stream>>>(
        p2, N2, p3, inv4,
        mW + 3L * C * C, mb + 3 * C, dW + 3L * 2 * C * C, db + 3 * C,
        dg + 3 * C, dbe + 3 * C, dm + 3 * C, dv + 3 * C, p2);
    dec_kernel<<<grows(N1), 256, 0, stream>>>(
        p1, N1, p2, inv2,
        mW + 4L * C * C, mb + 4 * C, dW + 4L * 2 * C * C, db + 4 * C,
        dg + 4 * C, dbe + 4 * C, dm + 4 * C, dv + 4 * C, p1);
}
